// MSP_2001454760189
// MI455X (gfx1250) — matrix operand order repaired and checked by reading, not compiled, not run on hardware
//
#include <hip/hip_runtime.h>

#define NPTS 500000
#define CCH  128
#define NVOX 65536

typedef _Float16 v16h __attribute__((ext_vector_type(16)));
typedef _Float16 v8h  __attribute__((ext_vector_type(8)));
typedef float    v8f  __attribute__((ext_vector_type(8)));
typedef float    v4f  __attribute__((ext_vector_type(4)));

// ---------------------------------------------------------------------------
// Stage 2: W (f32, [k][n] row-major) -> Wt (f16, [n][k] row-major)
// ---------------------------------------------------------------------------
__global__ void msp_prepw(const float* __restrict__ W1, const float* __restrict__ W2,
                          _Float16* __restrict__ W1t, _Float16* __restrict__ W2t) {
    int tid = blockIdx.x * 256 + threadIdx.x;          // 0 .. 16383
    int k = tid >> 7;
    int n = tid & 127;
    W1t[n * CCH + k] = (_Float16)W1[tid];
    W2t[n * CCH + k] = (_Float16)W2[tid];
}

// ---------------------------------------------------------------------------
// Stage 3: scatter-add feats into pooled[idx], count points per voxel
// one thread per (point, 4 channels)
// ---------------------------------------------------------------------------
__global__ void msp_scatter(const float* __restrict__ feats, const int* __restrict__ idx,
                            float* __restrict__ pooled, float* __restrict__ counts) {
    long tid = (long)blockIdx.x * blockDim.x + threadIdx.x;   // N*32 threads
    int i  = (int)(tid >> 5);
    int c4 = ((int)tid & 31) * 4;
    int v  = idx[i];
    const float* f = feats + (long)i * CCH + c4;
    float* p = pooled + (long)v * CCH + c4;
    atomicAdd(p + 0, f[0]);
    atomicAdd(p + 1, f[1]);
    atomicAdd(p + 2, f[2]);
    atomicAdd(p + 3, f[3]);
    if (c4 == 0) atomicAdd(counts + v, 1.0f);
}

// ---------------------------------------------------------------------------
// Stage 4: pooled /= max(counts, 1)
// ---------------------------------------------------------------------------
__global__ void msp_norm(float* __restrict__ pooled, const float* __restrict__ counts) {
    long tid = (long)blockIdx.x * blockDim.x + threadIdx.x;   // V*32 threads
    int v  = (int)(tid >> 5);
    int c4 = ((int)tid & 31) * 4;
    float inv = 1.0f / fmaxf(counts[v], 1.0f);
    float* p = pooled + (long)v * CCH + c4;
    p[0] *= inv; p[1] *= inv; p[2] *= inv; p[3] *= inv;
}

// ---------------------------------------------------------------------------
// Stage 5: fused  os = feats - pooled[idx];
//          h = relu(os@W1 + b1) * feats;  out = relu(h@W2 + b2)
// One wave (32 lanes) per 16-point tile; 8 waves / block.
// WMMA: v_wmma_f32_16x16x32_f16, f32 accumulate.
// ---------------------------------------------------------------------------
__global__ __launch_bounds__(256) void msp_main(
    const float* __restrict__ feats, const float* __restrict__ pooled,
    const int* __restrict__ idx, const float* __restrict__ b1,
    const float* __restrict__ b2, const _Float16* __restrict__ W1t,
    const _Float16* __restrict__ W2t, float* __restrict__ out, int numTiles) {

    __shared__ _Float16 hTile[8][16 * CCH];   // per-wave 16x128 f16 staging (32 KB)

    const int wave = threadIdx.x >> 5;
    const int lane = threadIdx.x & 31;
    const int tile = blockIdx.x * 8 + wave;
    if (tile >= numTiles) return;             // whole wave exits together (EXEC stays full)

    const int laneHi = lane >> 4;             // 0 or 1
    const int m      = lane & 15;             // row within tile / column within n-tile
    const long row   = (long)tile * 16 + m;
    const int  vox   = idx[row];
    const float* frow = feats  + row * CCH;
    const float* prow = pooled + (long)vox * CCH;

    // ---------------- GEMM 1: os @ W1 ----------------
    v8f acc[8];
#pragma unroll
    for (int t = 0; t < 8; ++t) acc[t] = 0.0f;

#pragma unroll
    for (int kt = 0; kt < 4; ++kt) {
        // A fragment (16-bit A 16x32 layout): lanes 0-15 rows, halves 0..7 = K k0..k0+7,
        // halves 8..15 = K k0+16..k0+23 ; lanes 16-31 same rows, K base +8.
        const int k0 = kt * 32 + laneHi * 8;
        v16h a;
#pragma unroll
        for (int g = 0; g < 2; ++g) {
            const int kk = k0 + g * 16;
            v4f f0 = *(const v4f*)(frow + kk);
            v4f f1 = *(const v4f*)(frow + kk + 4);
            v4f p0 = *(const v4f*)(prow + kk);
            v4f p1 = *(const v4f*)(prow + kk + 4);
            a[g * 8 + 0] = (_Float16)(f0.x - p0.x);
            a[g * 8 + 1] = (_Float16)(f0.y - p0.y);
            a[g * 8 + 2] = (_Float16)(f0.z - p0.z);
            a[g * 8 + 3] = (_Float16)(f0.w - p0.w);
            a[g * 8 + 4] = (_Float16)(f1.x - p1.x);
            a[g * 8 + 5] = (_Float16)(f1.y - p1.y);
            a[g * 8 + 6] = (_Float16)(f1.z - p1.z);
            a[g * 8 + 7] = (_Float16)(f1.w - p1.w);
        }
        // B fragment (32x16): lane = column n, halves 0..7 = K kb..kb+7, halves 8..15 = K kb+16..kb+23 (kb +8 for lanes 16-31)
        const int kb = kt * 32 + laneHi * 8;
#pragma unroll
        for (int t = 0; t < 8; ++t) {
            v16h b = __builtin_shufflevector(*(const v8h*)(W1t + (t * 16 + m) * CCH + kb), *(const v8h*)(W1t + (t * 16 + m) * CCH + kb + 16), 0, 1, 2, 3, 4, 5, 6, 7, 8, 9, 10, 11, 12, 13, 14, 15);
            acc[t] = __builtin_amdgcn_wmma_f32_16x16x32_f16(
                false, a, false, b, (short)0, acc[t], false, false);
        }
    }

    // ---------- epilogue 1: h = relu(z + b1) * feats, staged as f16 in LDS ----------
    _Float16* hbase = &hTile[wave][0];
#pragma unroll
    for (int t = 0; t < 8; ++t) {
        const int n = t * 16 + m;
        const float bias = b1[n];
#pragma unroll
        for (int j = 0; j < 8; ++j) {
            const int mm = j + laneHi * 8;            // C/D layout: VGPR j -> row j (+8 hi)
            float z = acc[t][j] + bias;
            z = z > 0.0f ? z : 0.0f;
            const float hv = z * feats[((long)tile * 16 + mm) * CCH + n];
            hbase[mm * CCH + n] = (_Float16)hv;
        }
    }
    // same-wave DS ops are in-order: loads below observe the stores above.

    // ---------------- GEMM 2: h @ W2 ----------------
    v8f acc2[8];
#pragma unroll
    for (int t = 0; t < 8; ++t) acc2[t] = 0.0f;

    const _Float16* hrow = &hTile[wave][m * CCH];
#pragma unroll
    for (int kt = 0; kt < 4; ++kt) {
        const int k0 = kt * 32 + laneHi * 8;
        v8h lo = *(const v8h*)(hrow + k0);
        v8h hi = *(const v8h*)(hrow + k0 + 16);
        v16h a = __builtin_shufflevector(lo, hi, 0, 1, 2, 3, 4, 5, 6, 7,
                                                 8, 9, 10, 11, 12, 13, 14, 15);
        const int kb = kt * 32 + laneHi * 8;
#pragma unroll
        for (int t = 0; t < 8; ++t) {
            v16h b = __builtin_shufflevector(*(const v8h*)(W2t + (t * 16 + m) * CCH + kb), *(const v8h*)(W2t + (t * 16 + m) * CCH + kb + 16), 0, 1, 2, 3, 4, 5, 6, 7, 8, 9, 10, 11, 12, 13, 14, 15);
            acc2[t] = __builtin_amdgcn_wmma_f32_16x16x32_f16(
                false, a, false, b, (short)0, acc2[t], false, false);
        }
    }

    // ---------- epilogue 2: out = relu(z + b2), f32 store ----------
#pragma unroll
    for (int t = 0; t < 8; ++t) {
        const int n = t * 16 + m;
        const float bias = b2[n];
#pragma unroll
        for (int j = 0; j < 8; ++j) {
            const int mm = j + laneHi * 8;
            float z = acc2[t][j] + bias;
            z = z > 0.0f ? z : 0.0f;
            out[((long)tile * 16 + mm) * CCH + n] = z;
        }
    }
}

// ---------------------------------------------------------------------------
extern "C" void kernel_launch(void* const* d_in, const int* in_sizes, int n_in,
                              void* d_out, int out_size, void* d_ws, size_t ws_size,
                              hipStream_t stream) {
    const float* feats = (const float*)d_in[0];
    const float* W1    = (const float*)d_in[1];
    const float* b1    = (const float*)d_in[2];
    const float* W2    = (const float*)d_in[3];
    const float* b2    = (const float*)d_in[4];
    const int*   idx   = (const int*)d_in[5];
    float*       out   = (float*)d_out;

    // workspace layout
    char* ws = (char*)d_ws;
    float*     pooled = (float*)ws;                                     // V*C f32
    float*     counts = (float*)(ws + (size_t)NVOX * CCH * 4);          // V   f32
    _Float16*  W1t    = (_Float16*)(ws + (size_t)NVOX * CCH * 4 + (size_t)NVOX * 4);
    _Float16*  W2t    = W1t + CCH * CCH;

    // 1) zero pooled + counts (contiguous)
    hipMemsetAsync(pooled, 0, ((size_t)NVOX * CCH + NVOX) * sizeof(float), stream);

    // 2) f16 transposed weights
    msp_prepw<<<64, 256, 0, stream>>>(W1, W2, W1t, W2t);

    // 3) scatter-add (N*32 threads)
    msp_scatter<<<(NPTS * 32) / 256, 256, 0, stream>>>(feats, idx, pooled, counts);

    // 4) normalize (V*32 threads)
    msp_norm<<<(NVOX * 32) / 256, 256, 0, stream>>>(pooled, counts);

    // 5) fused MLP with WMMA; N = 16 * 31250 exactly
    const int numTiles = NPTS / 16;
    msp_main<<<(numTiles + 7) / 8, 256, 0, stream>>>(feats, pooled, idx, b1, b2,
                                                     W1t, W2t, out, numTiles);
}
